// MERITS_63934883169129
// MI455X (gfx1250) — matrix operand order repaired and checked by reading, not compiled, not run on hardware
//
#include <hip/hip_runtime.h>
#include <hip/hip_bf16.h>
#include <math.h>

typedef __attribute__((ext_vector_type(16))) __bf16 v16bf;
typedef __attribute__((ext_vector_type(8)))  float  v8f;

static __device__ __forceinline__ unsigned short f2bf(float x) {
    union { float f; unsigned int u; } c; c.f = x;
    unsigned int u = c.u;
    u += 0x7fffu + ((u >> 16) & 1u);      // round-to-nearest-even
    return (unsigned short)(u >> 16);
}

static __device__ __forceinline__ unsigned int pack_bf16(float a, float b) {
#if __has_builtin(__builtin_amdgcn_cvt_pk_bf16_f32)
    typedef __attribute__((ext_vector_type(2))) __bf16 v2bf;
    union { v2bf v; unsigned int u; } r;
    r.v = __builtin_amdgcn_cvt_pk_bf16_f32(a, b);
    return r.u;
#else
    return (unsigned int)f2bf(a) | ((unsigned int)f2bf(b) << 16);
#endif
}

// ---------------------------------------------------------------------------
// Generic WMMA GEMM: out[M,N] = act(A')[M,K] @ W[K,N] + bias, optional relus.
// A row r maps to A[(r/aDiv)*aOuter + (r%aDiv)*aRow + k]  (handles med[:, :L]).
// Block = 256 threads = 8 waves; block tile 64x64: each wave owns two 16x16
// output tiles (rows mt*16 and mt*16+32) sharing one B fragment -> 4 v_wmma
// per barrier pair, and W is re-read only M/64 times.
// K-chunk = 64. Requires M%64, N%64, K%64 == 0.
// LDS rows are 33 dwords (odd stride => conflict-free staging + fragment reads).
// ---------------------------------------------------------------------------
__global__ void wmma_gemm(const float* __restrict__ A, const float* __restrict__ W,
                          const float* __restrict__ bias, float* __restrict__ out,
                          int M, int N, int K,
                          int aDiv, long long aOuter, long long aRow,
                          int reluA, int reluOut)
{
    __shared__ unsigned int As32[64 * 33];   // [m][kpair] bf16x2
    __shared__ unsigned int Bs32[64 * 33];   // [n][kpair] transposed bf16x2

    const int tid  = threadIdx.x;
    const int wave = tid >> 5;
    const int lane = tid & 31;
    const int half = lane >> 4;
    const int l16  = lane & 15;
    const int mt   = wave >> 2;              // 0..1 -> row tiles mt*16, mt*16+32
    const int nt   = wave & 3;               // 0..3
    const int blockM = blockIdx.y * 64;
    const int blockN = blockIdx.x * 64;

    v8f acc0 = {}, acc1 = {};
    for (int k0 = 0; k0 < K; k0 += 64) {
        // stage A chunk: 64 rows x 64 k  (2048 bf16 pairs, 8/thread)
        #pragma unroll
        for (int it = 0; it < 8; ++it) {
            int idx = tid + it * 256;
            int m  = idx >> 5;
            int kp = idx & 31;
            int row = blockM + m;
            long long off = (long long)(row / aDiv) * aOuter +
                            (long long)(row % aDiv) * aRow + (k0 + 2 * kp);
            float v0 = A[off], v1 = A[off + 1];
            if (reluA) { v0 = fmaxf(v0, 0.f); v1 = fmaxf(v1, 0.f); }
            As32[m * 33 + kp] = pack_bf16(v0, v1);
        }
        // stage W chunk transposed: Bs[n][k] = W[(k0+k)*N + blockN+n]  (2048 pairs)
        #pragma unroll
        for (int it = 0; it < 8; ++it) {
            int idx = tid + it * 256;
            int n  = idx & 63;
            int kp = idx >> 6;
            int k  = k0 + 2 * kp;
            float v0 = W[(long long)k * N + (blockN + n)];
            float v1 = W[(long long)(k + 1) * N + (blockN + n)];
            Bs32[n * 33 + kp] = pack_bf16(v0, v1);
        }
        __syncthreads();

        const unsigned int* ap0 = &As32[(mt * 16 + l16) * 33];
        const unsigned int* ap1 = &As32[(32 + mt * 16 + l16) * 33];
        const unsigned int* bp  = &Bs32[(nt * 16 + l16) * 33];
        #pragma unroll
        for (int s = 0; s < 2; ++s) {
            union { v16bf v; unsigned int u[8]; } a0, a1, bb;
            #pragma unroll
            for (int v = 0; v < 4; ++v) {
                // A 16-bit layout: vgpr v: K=8*half+2v,+1 ; vgpr v+4: K=16+8*half+2v,+1
                a0.u[v]     = ap0[16 * s + 4 * half + v];
                a0.u[v + 4] = ap0[16 * s + 8 + 4 * half + v];
                a1.u[v]     = ap1[16 * s + 4 * half + v];
                a1.u[v + 4] = ap1[16 * s + 8 + 4 * half + v];
            }
            #pragma unroll
            for (int v = 0; v < 8; ++v) {
                // B 16-bit layout: same as A: vgpr v<4: K=8*half+2v,+1 ; v>=4: K=16+8*half+2(v-4),+1
                bb.u[v] = bp[16 * s + 4 * half + v + 4 * (v >> 2)];
            }
            acc0 = __builtin_amdgcn_wmma_f32_16x16x32_bf16(false, a0.v, false, bb.v,
                                                           (short)0, acc0, false, false);
            acc1 = __builtin_amdgcn_wmma_f32_16x16x32_bf16(false, a1.v, false, bb.v,
                                                           (short)0, acc1, false, false);
        }
        __syncthreads();
    }

    const int col = blockN + nt * 16 + l16;
    const float bv = bias ? bias[col] : 0.f;
    #pragma unroll
    for (int r = 0; r < 8; ++r) {
        int row0 = blockM + mt * 16 + half * 8 + r;  // C/D: M = vgpr + 8*half
        float v0 = acc0[r] + bv;
        if (reluOut) v0 = fmaxf(v0, 0.f);
        out[(long long)row0 * N + col] = v0;
        int row1 = row0 + 32;
        float v1 = acc1[r] + bv;
        if (reluOut) v1 = fmaxf(v1, 0.f);
        out[(long long)row1 * N + col] = v1;
    }
}

// ---------------------------------------------------------------------------
// static = relu(relu(lab@sW1+sb1)@sW2+sb2)   [B,32]
// ---------------------------------------------------------------------------
__global__ void static_kernel(const float* __restrict__ lab,
                              const float* __restrict__ sW1, const float* __restrict__ sb1,
                              const float* __restrict__ sW2, const float* __restrict__ sb2,
                              float* __restrict__ statf)
{
    const int b = blockIdx.x, t = threadIdx.x;           // 64 threads
    __shared__ float lr[128], s1[64];
    lr[t] = lab[b * 128 + t];
    lr[t + 64] = lab[b * 128 + 64 + t];
    __syncthreads();
    float a = sb1[t];
    for (int k = 0; k < 128; ++k) a += lr[k] * sW1[k * 64 + t];
    s1[t] = a > 0.f ? a : 0.f;
    __syncthreads();
    if (t < 32) {
        float c = sb2[t];
        for (int k = 0; k < 64; ++k) c += s1[k] * sW2[k * 32 + t];
        statf[b * 32 + t] = c > 0.f ? c : 0.f;
    }
}

// ---------------------------------------------------------------------------
// hsum[b,n] = sum_l<24 med[b,l,n]
// ---------------------------------------------------------------------------
__global__ void hsum_kernel(const float* __restrict__ med, float* __restrict__ hsum)
{
    const int b = blockIdx.x, n = threadIdx.x;           // 256 threads
    float s = 0.f;
    for (int l = 0; l < 24; ++l) s += med[((long long)b * 25 + l) * 256 + n];
    hsum[b * 256 + n] = s;
}

// ---------------------------------------------------------------------------
// 3 GATs fused. Since Wh = h@W (h:[B,N,1], W:[1,E]), each GAT collapses to a
// masked softmax over rows with scalar logits, output s_i * W[e].
// ---------------------------------------------------------------------------
static __device__ __forceinline__
float gat_row(const float* hh, float hi, const float* adj_row,
              const float* W, const float* a1, const float* a2)
{
    float c1 = 0.f, c2 = 0.f;
    for (int e = 0; e < 64; ++e) { c1 += W[e] * a1[e]; c2 += W[e] * a2[e]; }
    const float base = hi * c1;
    float m = -1e30f;
    for (int j = 0; j < 256; ++j) {
        if (adj_row[j] > 0.f) {
            float e = base + hh[j] * c2;
            e = e >= 0.f ? e : 0.2f * e;             // leaky_relu 0.2
            m = fmaxf(m, e);
        }
    }
    float se = 0.f, sa = 0.f;
    for (int j = 0; j < 256; ++j) {
        if (adj_row[j] > 0.f) {
            float e = base + hh[j] * c2;
            e = e >= 0.f ? e : 0.2f * e;
            float p = __expf(e - m);
            se += p; sa += p * hh[j];
        }
    }
    return sa / se;
}

__global__ void gat_kernel(const float* __restrict__ hsum,
                           const float* __restrict__ ehr, const float* __restrict__ tim,
                           const float* __restrict__ ddi,
                           const float* __restrict__ geW, const float* __restrict__ gea1,
                           const float* __restrict__ gea2,
                           const float* __restrict__ gtW, const float* __restrict__ gta1,
                           const float* __restrict__ gta2,
                           const float* __restrict__ gdW, const float* __restrict__ gda1,
                           const float* __restrict__ gda2,
                           float* __restrict__ drug_mem)
{
    const int b = blockIdx.x, i = threadIdx.x;           // 256 threads
    __shared__ float hh[256];
    __shared__ float s0[256], s1[256], s2[256];
    hh[i] = hsum[b * 256 + i];
    __syncthreads();
    const float hi = hh[i];
    s0[i] = gat_row(hh, hi, ehr + i * 256, geW, gea1, gea2);
    s1[i] = gat_row(hh, hi, tim + i * 256, gtW, gta1, gta2);
    s2[i] = gat_row(hh, hi, ddi + i * 256, gdW, gda1, gda2);
    __syncthreads();
    for (int rep = 0; rep < 64; ++rep) {                 // coalesced writes
        int flat = rep * 256 + i;
        int n = flat >> 6, e = flat & 63;
        drug_mem[((long long)b * 256 + n) * 64 + e] =
            s0[n] * geW[e] + s1[n] * gtW[e] - s2[n] * gdW[e];
    }
}

// ---------------------------------------------------------------------------
// GRU over T steps, hidden 16; keep h at step length-1 (run only len steps).
// One 64-thread block per (b,l) sequence.
// ---------------------------------------------------------------------------
__global__ void gru_kernel(const float* __restrict__ glu, const float* __restrict__ tim,
                           const int* __restrict__ length,
                           const float* __restrict__ Wih, const float* __restrict__ Whh,
                           const float* __restrict__ bih, const float* __restrict__ bhh,
                           float* __restrict__ glu_enc)
{
    const int seq = blockIdx.x;                          // b*24 + l
    const int tid = threadIdx.x;                         // 64 threads
    __shared__ float h[16], xs[17], gi[48], gh[48];
    if (tid < 16) h[tid] = 0.f;
    int len = length[seq];
    if (len < 1) len = 1; if (len > 25) len = 25;
    __syncthreads();
    for (int t = 0; t < len; ++t) {
        if (tid < 16)       xs[tid] = glu[((long long)seq * 25 + t) * 16 + tid];
        else if (tid == 16) xs[16]  = tim[(long long)seq * 25 + t];
        __syncthreads();
        if (tid < 48) {
            float a = bih[tid];
            for (int i = 0; i < 17; ++i) a += xs[i] * Wih[i * 48 + tid];
            gi[tid] = a;
            float c = bhh[tid];
            for (int j = 0; j < 16; ++j) c += h[j] * Whh[j * 48 + tid];
            gh[tid] = c;
        }
        __syncthreads();
        if (tid < 16) {
            float r = 1.f / (1.f + __expf(-(gi[tid] + gh[tid])));
            float z = 1.f / (1.f + __expf(-(gi[16 + tid] + gh[16 + tid])));
            float n = tanhf(gi[32 + tid] + r * gh[32 + tid]);
            h[tid] = (1.f - z) * n + z * h[tid];
        }
        __syncthreads();
    }
    if (tid < 16) glu_enc[(long long)seq * 16 + tid] = h[tid];
}

// ---------------------------------------------------------------------------
// med_rep gating: med_rep = hfeat * sigmoid(tanh(hfeat@mWa)@mua)
// ---------------------------------------------------------------------------
__global__ void medrep_kernel(const float* __restrict__ hfeat,
                              const float* __restrict__ mWa, const float* __restrict__ mua,
                              float* __restrict__ med_rep)
{
    const int row = blockIdx.x, e = threadIdx.x;         // 64 threads
    __shared__ float hf[64], tt[32];
    hf[e] = hfeat[(long long)row * 64 + e];
    __syncthreads();
    if (e < 32) {
        float s = 0.f;
        for (int k = 0; k < 64; ++k) s += hf[k] * mWa[k * 32 + e];
        tt[e] = tanhf(s);
    }
    __syncthreads();
    float asum = 0.f;
    for (int j = 0; j < 32; ++j) asum += tt[j] * mua[j];
    float a = 1.f / (1.f + __expf(-asum));
    med_rep[(long long)row * 64 + e] = hf[e] * a;
}

// ---------------------------------------------------------------------------
// glu attn-enc + concat static -> patient_rep[B,L,64]
// ---------------------------------------------------------------------------
__global__ void patient_kernel(const float* __restrict__ glu_enc,
                               const float* __restrict__ statf,
                               const float* __restrict__ gWin, const float* __restrict__ gbin,
                               const float* __restrict__ gWa,  const float* __restrict__ gua,
                               float* __restrict__ patient)
{
    const int row = blockIdx.x, e = threadIdx.x;         // 32 threads
    const int b = row / 24;
    __shared__ float ge[16], hf[32], tt[16];
    if (e < 16) ge[e] = glu_enc[(long long)row * 16 + e];
    __syncthreads();
    float s = gbin[e];
    for (int k = 0; k < 16; ++k) s += ge[k] * gWin[k * 32 + e];
    hf[e] = s > 0.f ? s : 0.f;
    __syncthreads();
    if (e < 16) {
        float u = 0.f;
        for (int k = 0; k < 32; ++k) u += hf[k] * gWa[k * 16 + e];
        tt[e] = tanhf(u);
    }
    __syncthreads();
    float asum = 0.f;
    for (int j = 0; j < 16; ++j) asum += tt[j] * gua[j];
    float a = 1.f / (1.f + __expf(-asum));
    patient[(long long)row * 64 + e] = hf[e] * a;
    patient[(long long)row * 64 + 32 + e] = statf[b * 32 + e];
}

// ---------------------------------------------------------------------------
// MHA cores: 4 heads of dh=16, 24 keys. One block per (batch, head).
// ---------------------------------------------------------------------------
__global__ void attn1_kernel(const float* __restrict__ Q, const float* __restrict__ Kx,
                             const float* __restrict__ Vx, float* __restrict__ O)
{
    const int b = blockIdx.x >> 2, hd = blockIdx.x & 3;
    const int t = threadIdx.x;                           // 32 threads
    __shared__ float Ks[24][16], Vs[24][16];
    for (int idx = t; idx < 384; idx += 32) {
        int j = idx >> 4, c = idx & 15;
        Ks[j][c] = Kx[((long long)b * 24 + j) * 64 + hd * 16 + c];
        Vs[j][c] = Vx[((long long)b * 24 + j) * 64 + hd * 16 + c];
    }
    __syncthreads();
    if (t < 24) {
        float q[16];
        for (int c = 0; c < 16; ++c) q[c] = Q[((long long)b * 24 + t) * 64 + hd * 16 + c];
        float s[24]; float m = -1e30f;
        for (int j = 0; j < 24; ++j) {
            float d = 0.f;
            for (int c = 0; c < 16; ++c) d += q[c] * Ks[j][c];
            d *= 0.25f;                                  // 1/sqrt(16)
            s[j] = d; m = fmaxf(m, d);
        }
        float se = 0.f;
        for (int j = 0; j < 24; ++j) { s[j] = __expf(s[j] - m); se += s[j]; }
        const float inv = 1.f / se;
        float acc[16];
        for (int c = 0; c < 16; ++c) acc[c] = 0.f;
        for (int j = 0; j < 24; ++j) {
            float p = s[j] * inv;
            for (int c = 0; c < 16; ++c) acc[c] += p * Vs[j][c];
        }
        for (int c = 0; c < 16; ++c)
            O[((long long)b * 24 + t) * 64 + hd * 16 + c] = acc[c];
    }
}

__global__ void attn2_kernel(const float* __restrict__ Q, const float* __restrict__ Kx,
                             const float* __restrict__ Vx, float* __restrict__ O)
{
    const int b = blockIdx.x >> 2, hd = blockIdx.x & 3;
    const int t = threadIdx.x;                           // 256 threads (queries)
    __shared__ float Ks[24][16], Vs[24][16];
    for (int idx = t; idx < 384; idx += 256) {
        int j = idx >> 4, c = idx & 15;
        Ks[j][c] = Kx[((long long)b * 24 + j) * 64 + hd * 16 + c];
        Vs[j][c] = Vx[((long long)b * 24 + j) * 64 + hd * 16 + c];
    }
    __syncthreads();
    float q[16];
    for (int c = 0; c < 16; ++c) q[c] = Q[((long long)b * 256 + t) * 64 + hd * 16 + c];
    float s[24]; float m = -1e30f;
    for (int j = 0; j < 24; ++j) {
        float d = 0.f;
        for (int c = 0; c < 16; ++c) d += q[c] * Ks[j][c];
        d *= 0.25f;
        s[j] = d; m = fmaxf(m, d);
    }
    float se = 0.f;
    for (int j = 0; j < 24; ++j) { s[j] = __expf(s[j] - m); se += s[j]; }
    const float inv = 1.f / se;
    float acc[16];
    for (int c = 0; c < 16; ++c) acc[c] = 0.f;
    for (int j = 0; j < 24; ++j) {
        float p = s[j] * inv;
        for (int c = 0; c < 16; ++c) acc[c] += p * Vs[j][c];
    }
    for (int c = 0; c < 16; ++c)
        O[((long long)b * 256 + t) * 64 + hd * 16 + c] = acc[c];
}

// ---------------------------------------------------------------------------
extern "C" void kernel_launch(void* const* d_in, const int* in_sizes, int n_in,
                              void* d_out, int out_size, void* d_ws, size_t ws_size,
                              hipStream_t stream)
{
    (void)in_sizes; (void)n_in; (void)out_size; (void)ws_size;
    // inputs in setup_inputs() dict order; params dict flattened in insertion order
    const float* lab   = (const float*)d_in[0];
    const float* glu   = (const float*)d_in[1];
    const float* tim   = (const float*)d_in[2];
    const float* med   = (const float*)d_in[3];
    const int*   len   = (const int*)  d_in[4];
    const float* ehr   = (const float*)d_in[5];
    const float* ddi   = (const float*)d_in[6];
    const float* tadj  = (const float*)d_in[7];
    const float* sW1   = (const float*)d_in[8];
    const float* sb1   = (const float*)d_in[9];
    const float* sW2   = (const float*)d_in[10];
    const float* sb2   = (const float*)d_in[11];
    const float* mWin  = (const float*)d_in[12];
    const float* mbin  = (const float*)d_in[13];
    const float* mWa   = (const float*)d_in[14];
    const float* mua   = (const float*)d_in[15];
    const float* gWin  = (const float*)d_in[16];
    const float* gbin  = (const float*)d_in[17];
    const float* gWa   = (const float*)d_in[18];
    const float* gua   = (const float*)d_in[19];
    const float* gruWih= (const float*)d_in[20];
    const float* gruWhh= (const float*)d_in[21];
    const float* grubih= (const float*)d_in[22];
    const float* grubhh= (const float*)d_in[23];
    const float* geW   = (const float*)d_in[24];
    const float* gea1  = (const float*)d_in[25];
    const float* gea2  = (const float*)d_in[26];
    const float* gtW   = (const float*)d_in[27];
    const float* gta1  = (const float*)d_in[28];
    const float* gta2  = (const float*)d_in[29];
    const float* gdW   = (const float*)d_in[30];
    const float* gda1  = (const float*)d_in[31];
    const float* gda2  = (const float*)d_in[32];
    const float* m1Wq  = (const float*)d_in[33];
    const float* m1bq  = (const float*)d_in[34];
    const float* m1Wk  = (const float*)d_in[35];
    const float* m1bk  = (const float*)d_in[36];
    const float* m1Wv  = (const float*)d_in[37];
    const float* m1bv  = (const float*)d_in[38];
    const float* m1Wo  = (const float*)d_in[39];
    const float* m1bo  = (const float*)d_in[40];
    const float* m2Wq  = (const float*)d_in[41];
    const float* m2bq  = (const float*)d_in[42];
    const float* m2Wk  = (const float*)d_in[43];
    const float* m2bk  = (const float*)d_in[44];
    const float* m2Wv  = (const float*)d_in[45];
    const float* m2bv  = (const float*)d_in[46];
    const float* m2Wo  = (const float*)d_in[47];
    const float* m2bo  = (const float*)d_in[48];
    const float* oW1   = (const float*)d_in[49];
    const float* ob1   = (const float*)d_in[50];
    const float* oW2   = (const float*)d_in[51];
    const float* ob2   = (const float*)d_in[52];
    float* out = (float*)d_out;

    float* ws = (float*)d_ws;
    size_t off = 0;
    auto alloc = [&](size_t n) { float* p = ws + off; off += n; return p; };

    const int B = 256, L = 24, NMED = 256, E = 64;
    float* statf    = alloc((size_t)B * 32);
    float* hsum     = alloc((size_t)B * NMED);
    float* drug_mem = alloc((size_t)B * NMED * E);       // reused later as attnO2
    float* hfeat    = alloc((size_t)B * L * E);
    float* med_rep  = alloc((size_t)B * L * E);
    float* glu_enc  = alloc((size_t)B * L * 16);
    float* patient  = alloc((size_t)B * L * E);
    float* Q1       = alloc((size_t)B * L * E);
    float* K1       = alloc((size_t)B * L * E);
    float* V1       = alloc((size_t)B * L * E);
    float* O1       = alloc((size_t)B * L * E);
    float* Een      = alloc((size_t)B * L * E);
    float* Q2       = alloc((size_t)B * NMED * E);       // reused later as E_de
    float* K2       = alloc((size_t)B * L * E);
    float* V2       = alloc((size_t)B * L * E);
    float* hidden   = alloc((size_t)B * 2048);

    auto gemm = [&](const float* A, const float* W, const float* bias, float* o,
                    int M, int N, int K, int aDiv, long long aOuter, long long aRow,
                    int reluA, int reluOut) {
        dim3 grid(N / 64, M / 64);
        wmma_gemm<<<grid, 256, 0, stream>>>(A, W, bias, o, M, N, K,
                                            aDiv, aOuter, aRow, reluA, reluOut);
    };

    // 1. static lab features
    static_kernel<<<B, 64, 0, stream>>>(lab, sW1, sb1, sW2, sb2, statf);
    // 2. summed drug history
    hsum_kernel<<<B, 256, 0, stream>>>(med, hsum);
    // 3. fused 3-GAT drug memory
    gat_kernel<<<B, 256, 0, stream>>>(hsum, ehr, tadj, ddi,
                                      geW, gea1, gea2, gtW, gta1, gta2,
                                      gdW, gda1, gda2, drug_mem);
    // 4. GRU glucose encoder (stops at length-1)
    gru_kernel<<<B * L, 64, 0, stream>>>(glu, tim, len, gruWih, gruWhh,
                                         grubih, grubhh, glu_enc);
    // 5. med encoder hfeat = relu(med[:, :L] @ mWin + mbin)  (A-row remap skips visit 24)
    gemm(med, mWin, mbin, hfeat, B * L, 64, 256, 24, 25LL * 256, 256, 0, 1);
    // 6. med_rep gating
    medrep_kernel<<<B * L, 64, 0, stream>>>(hfeat, mWa, mua, med_rep);
    // 7. glu attn-enc + concat static -> patient_rep
    patient_kernel<<<B * L, 32, 0, stream>>>(glu_enc, statf, gWin, gbin, gWa, gua, patient);
    // 8. MHA1 projections (query=med_rep, key/value=patient_rep)
    gemm(med_rep, m1Wq, m1bq, Q1, B * L, 64, 64, B * L, 0, 64, 0, 0);
    gemm(patient, m1Wk, m1bk, K1, B * L, 64, 64, B * L, 0, 64, 0, 0);
    gemm(patient, m1Wv, m1bv, V1, B * L, 64, 64, B * L, 0, 64, 0, 0);
    attn1_kernel<<<B * 4, 32, 0, stream>>>(Q1, K1, V1, O1);
    gemm(O1, m1Wo, m1bo, Een, B * L, 64, 64, B * L, 0, 64, 0, 0);
    // 9. MHA2 projections (query=drug_mem, key/value=E_en)
    gemm(drug_mem, m2Wq, m2bq, Q2, B * NMED, 64, 64, B * NMED, 0, 64, 0, 0);
    gemm(Een, m2Wk, m2bk, K2, B * L, 64, 64, B * L, 0, 64, 0, 0);
    gemm(Een, m2Wv, m2bv, V2, B * L, 64, 64, B * L, 0, 64, 0, 0);
    float* O2  = drug_mem;   // drug_mem no longer needed
    attn2_kernel<<<B * 4, 256, 0, stream>>>(Q2, K2, V2, O2);
    float* Ede = Q2;         // Q2 no longer needed
    gemm(O2, m2Wo, m2bo, Ede, B * NMED, 64, 64, B * NMED, 0, 64, 0, 0);
    // 10. output MLP: hidden = relu(relu(Ede_flat[256,16384]) @ oW1 + ob1)
    gemm(Ede, oW1, ob1, hidden, B, 2048, 16384, B, 0, 16384, 1, 1);
    //     out = hidden @ oW2 + ob2
    gemm(hidden, oW2, ob2, out, B, 256, 2048, B, 0, 2048, 0, 0);
}
